// UNetTransformerBlock_78546361909339
// MI455X (gfx1250) — matrix operand order repaired and checked by reading, not compiled, not run on hardware
//
#include <hip/hip_runtime.h>
#include <math.h>

#define HH 256
#define WW 256
#define HW (HH*WW)
#define NB 2

typedef __attribute__((ext_vector_type(16))) _Float16 v16h;
typedef __attribute__((ext_vector_type(8)))  _Float16 v8h;
typedef __attribute__((ext_vector_type(8)))  float    v8f;

__device__ __forceinline__ int imin(int a, int b) { return a < b ? a : b; }

// ---- WMMA fragment builders (CDNA5 16x16x32 f16, wave32) ----
// A 16x32 f16: lanes 0-15 -> M=lane, K {0..7,16..23}; lanes 16-31 -> M=lane-16, K+8.
// Halves 0..7 are K-contiguous, halves 8..15 are K-contiguous -> two ds_load_b128.
__device__ __forceinline__ v16h frag_a_rowmajor(const _Float16* A, int ld, int m0, int k0) {
  const int lane = threadIdx.x & 31;
  const int m = m0 + (lane & 15);
  const int khi = (lane >= 16) ? 8 : 0;
  const _Float16* base = A + (long)m * ld + k0 + khi;
  const v8h lo = *(const v8h*)(base);
  const v8h hi = *(const v8h*)(base + 16);
  v16h f;
#pragma unroll
  for (int h = 0; h < 8; ++h) { f[h] = lo[h]; f[h + 8] = hi[h]; }
  return f;
}
// B 32x16 f16, source stored N-major: B(k,n) = S[n*ld + k].
// Same K order as A: K {0..7,16..23}, lanes 16-31 K+8 -> two ds_load_b128.
__device__ __forceinline__ v16h frag_b_nmajor(const _Float16* S, int ld, int k0, int n0) {
  const int lane = threadIdx.x & 31;
  const int n = n0 + (lane & 15);
  const int khi = (lane >= 16) ? 8 : 0;
  const _Float16* base = S + (long)n * ld + k0 + khi;
  const v8h lo = *(const v8h*)(base);
  const v8h hi = *(const v8h*)(base + 16);
  v16h f;
#pragma unroll
  for (int h = 0; h < 8; ++h) { f[h] = lo[h]; f[h + 8] = hi[h]; }
  return f;
}

// ---- bilinear align-corners resize (NCHW) ----
__global__ void resize_ac_kernel(const float* __restrict__ X, float* __restrict__ Y,
                                 int C, int hs, int wsrc, int hd, int wd) {
  const long total = (long)NB * C * hd * wd;
  const float sy = (hd > 1) ? (float)(hs - 1) / (float)(hd - 1) : 0.f;
  const float sx = (wd > 1) ? (float)(wsrc - 1) / (float)(wd - 1) : 0.f;
  for (long i = (long)blockIdx.x * blockDim.x + threadIdx.x; i < total;
       i += (long)gridDim.x * blockDim.x) {
    const int x = (int)(i % wd); long t = i / wd;
    const int y = (int)(t % hd); t /= hd;
    const int c = (int)(t % C);  const int n = (int)(t / C);
    const float fy = y * sy; const int y0 = (int)floorf(fy);
    const int y1 = imin(y0 + 1, hs - 1); const float wy = fy - (float)y0;
    const float fx = x * sx; const int x0 = (int)floorf(fx);
    const int x1 = imin(x0 + 1, wsrc - 1); const float wx = fx - (float)x0;
    const float* p = X + ((long)n * C + c) * hs * wsrc;
    const float v00 = p[y0*wsrc + x0], v01 = p[y0*wsrc + x1];
    const float v10 = p[y1*wsrc + x0], v11 = p[y1*wsrc + x1];
    Y[i] = v00*(1.f-wy)*(1.f-wx) + v01*(1.f-wy)*wx + v10*wy*(1.f-wx) + v11*wy*wx;
  }
}

// ---- per-channel BN stats -> folded scale/shift ----
__global__ void bn_stats_kernel(const float* __restrict__ X, int C, int hw,
                                const float* __restrict__ g, const float* __restrict__ b,
                                float* __restrict__ scale, float* __restrict__ shift,
                                int outoff) {
  const int c = blockIdx.x;
  __shared__ float ssum[256], ssq[256];
  float s = 0.f, q = 0.f;
  for (int n = 0; n < NB; ++n) {
    const float* p = X + ((long)n * C + c) * hw;
    for (int i = threadIdx.x; i < hw; i += 256) {
      const float v = p[i]; s += v; q += v * v;
    }
  }
  ssum[threadIdx.x] = s; ssq[threadIdx.x] = q;
  __syncthreads();
  for (int o = 128; o > 0; o >>= 1) {
    if ((int)threadIdx.x < o) {
      ssum[threadIdx.x] += ssum[threadIdx.x + o];
      ssq[threadIdx.x]  += ssq[threadIdx.x + o];
    }
    __syncthreads();
  }
  if (threadIdx.x == 0) {
    const float cnt = (float)NB * (float)hw;
    const float mean = ssum[0] / cnt;
    const float var  = ssq[0] / cnt - mean * mean;
    const float sc = g[c] * rsqrtf(var + 1e-5f);
    scale[outoff + c] = sc;
    shift[outoff + c] = b[c] - mean * sc;
  }
}

// ---- generic 1x1 conv with folded BN, as pixel-GEMM via WMMA ----
// HWSZ compile-time => pixel->(n,sp) split is shift/mask.
template <int HWSZ>
__global__ void conv1x1_bn_kernel(const float* __restrict__ X0, const float* __restrict__ X1,
                                  int c_split, int Cin,
                                  const float* __restrict__ scale, const float* __restrict__ shift,
                                  const float* __restrict__ Wt, const float* __restrict__ bias,
                                  int Cout, int cstride, float* __restrict__ Y, int relu) {
  __shared__ __align__(16) _Float16 As[64 * 32];   // [pixel][k]   (row-major A)
  __shared__ __align__(16) _Float16 Bs[32 * 32];   // [out][k]     (N-major B)
  const int tid = threadIdx.x;
  const int lane = tid & 31;
  const int wave = tid >> 5;
  const int mt = wave & 3;
  const int nt = wave >> 2;
  const long Mbase = (long)blockIdx.x * 64;
  const int Nbase = blockIdx.y * 32;
  v8f acc = {};
  for (int k0 = 0; k0 < Cin; k0 += 32) {
    if (k0 + 32 < Cin)
      __builtin_prefetch(&Wt[(long)Nbase * Cin + k0 + 32], 0, 2);  // global_prefetch_b8
    // stage A: consecutive threads -> consecutive pixels (coalesced global reads)
    for (int idx = tid; idx < 64 * 32; idx += 256) {
      const int mp = idx & 63; const int kc = idx >> 6;
      const long p = Mbase + mp;
      const int n = (int)(p / HWSZ); const int sp = (int)(p % HWSZ);
      const int c = k0 + kc;
      float v;
      if (c < c_split) v = X0[((long)n * c_split + c) * HWSZ + sp];
      else             v = X1[((long)n * (Cin - c_split) + (c - c_split)) * HWSZ + sp];
      v = v * scale[c] + shift[c];
      As[mp * 32 + kc] = (_Float16)v;
    }
    // stage B N-major: consecutive threads -> consecutive k (coalesced weight reads)
    for (int idx = tid; idx < 32 * 32; idx += 256) {
      const int kc = idx & 31; const int on = idx >> 5;
      const int o = Nbase + on;
      const float v = (o < Cout) ? Wt[(long)o * Cin + k0 + kc] : 0.f;
      Bs[on * 32 + kc] = (_Float16)v;
    }
    __syncthreads();
    const v16h a = frag_a_rowmajor(As, 32, mt * 16, 0);
    const v16h b = frag_b_nmajor(Bs, 32, 0, nt * 16);
    acc = __builtin_amdgcn_wmma_f32_16x16x32_f16(false, a, false, b, (short)0, acc, false, false);
    __syncthreads();
  }
  const int nn = Nbase + nt * 16 + (lane & 15);
  if (nn < Cout) {
    const float bval = bias[nn];
    const int hi8 = (lane >= 16) ? 8 : 0;
#pragma unroll
    for (int r = 0; r < 8; ++r) {
      const long p = Mbase + mt * 16 + r + hi8;
      const int n = (int)(p / HWSZ); const int sp = (int)(p % HWSZ);
      float v = acc[r] + bval;
      if (relu) v = fmaxf(v, 0.f);
      Y[((long)n * cstride + nn) * HWSZ + sp] = v;
    }
  }
}

// ---- grid_sample with border clamp (temporal warp). resize-to-same-size is identity. ----
__global__ void warp_kernel(const float* __restrict__ tkv, const float* __restrict__ mv,
                            float* __restrict__ out) {
  const long total = (long)NB * HW;
  for (long i = (long)blockIdx.x * blockDim.x + threadIdx.x; i < total;
       i += (long)gridDim.x * blockDim.x) {
    const int sp = (int)(i % HW); const int n = (int)(i / HW);
    const int y = sp / WW, x = sp % WW;
    const float xs = -1.f + 2.f * (float)x / (float)(WW - 1);
    const float ys = -1.f + 2.f * (float)y / (float)(HH - 1);
    const float gx = mv[((long)n * 2 + 0) * HW + sp] * (2.f / WW) + xs;
    const float gy = ys - mv[((long)n * 2 + 1) * HW + sp] * (2.f / HH);
    const float ix = fminf(fmaxf((gx + 1.f) * 0.5f * (WW - 1), 0.f), (float)(WW - 1));
    const float iy = fminf(fmaxf((gy + 1.f) * 0.5f * (HH - 1), 0.f), (float)(HH - 1));
    const int x0 = (int)floorf(ix), y0 = (int)floorf(iy);
    const int x1 = imin(x0 + 1, WW - 1), y1 = imin(y0 + 1, HH - 1);
    const float wx = ix - (float)x0, wy = iy - (float)y0;
    const float w00 = (1.f-wy)*(1.f-wx), w01 = (1.f-wy)*wx, w10 = wy*(1.f-wx), w11 = wy*wx;
    const float* base = tkv + (long)n * 128 * HW;
    for (int c = 0; c < 128; ++c) {
      const float* p = base + (long)c * HW;
      out[((long)n * 128 + c) * HW + sp] =
          p[y0*WW + x0]*w00 + p[y0*WW + x1]*w01 + p[y1*WW + x0]*w10 + p[y1*WW + x1]*w11;
    }
  }
}

// ---- channel-range copy (NCHW) ----
__global__ void chcopy_kernel(const float* __restrict__ src, int scs, int sco,
                              float* __restrict__ dst, int dcs, int dco, int C, int hw) {
  const long total = (long)NB * C * hw;
  for (long i = (long)blockIdx.x * blockDim.x + threadIdx.x; i < total;
       i += (long)gridDim.x * blockDim.x) {
    const int sp = (int)(i % hw); long t = i / hw;
    const int c = (int)(t % C); const int n = (int)(t / C);
    dst[((long)n * dcs + dco + c) * hw + sp] = src[((long)n * scs + sco + c) * hw + sp];
  }
}

// ---- flash windowed attention: one block per (n, window); 8 waves ----
// Q: 64 tokens x 64 ch; KV: 384 tokens (256 band + 64 temporal + 64 ds) in 6 chunks of 64.
// Staging buffer channel-major [c][t]; scores stored transposed [kvtok][qtok].
__global__ void attn_kernel(const float* __restrict__ su_emb,   // (N,189,H,W)
                            const float* __restrict__ radiance, // (N,3,H,W)
                            const float* __restrict__ warped,   // (N,128,H,W)
                            const float* __restrict__ dsemb,    // (N,128,H,W)
                            const float* __restrict__ qg, const float* __restrict__ qb,
                            const float* __restrict__ kg, const float* __restrict__ kb,
                            const float* __restrict__ vg, const float* __restrict__ vb,
                            float* __restrict__ img_attn) {     // (N,64,H,W)
  __shared__ __align__(16) _Float16 Q16[64 * 64];   // [qtok][ch]     A for S
  __shared__ __align__(16) _Float16 K16[64 * 64];   // [kvtok][ch]    B (N-major) for S; reused as P [qtok][kvtok]
  __shared__ __align__(16) _Float16 V16t[64 * 64];  // [ch][kvtok]    B (N-major) for O
  __shared__ __align__(32) float    Sbuf[64 * 64];  // staging [c][t] / scores [kvtok][qtok]
  __shared__ float mrow[64], lrow[64], arow[64];

  const int tid = threadIdx.x;
  const int lane = tid & 31;
  const int wave = tid >> 5;
  const int blk = blockIdx.x;
  const int n = blk >> 10;
  const int l = blk & 1023;
  const int lh = l >> 5, lw = l & 31;
  const int y0 = lh * 8, x0 = lw * 8;
  const int hi8 = (lane >= 16) ? 8 : 0;

  // ---- Q: stage (coalesced: consecutive threads -> consecutive tokens), LN, fold 1/8 ----
  for (int idx = tid; idx < 64 * 64; idx += 256) {
    const int t = idx & 63, c = idx >> 6;
    const int sp = (y0 + (t >> 3)) * WW + (x0 + (t & 7));
    Sbuf[c * 64 + t] = su_emb[((long)n * 189 + c) * HW + sp];
  }
  __syncthreads();
  if (tid < 64) {
    float s = 0.f, q = 0.f;
    for (int c = 0; c < 64; ++c) { const float v = Sbuf[c*64 + tid]; s += v; q += v*v; }
    const float mean = s * (1.f/64.f);
    const float r = rsqrtf(q * (1.f/64.f) - mean*mean + 1e-5f);
    for (int c = 0; c < 64; ++c)
      Q16[tid*64 + c] = (_Float16)(((Sbuf[c*64 + tid] - mean) * r * qg[c] + qb[c]) * 0.125f);
    mrow[tid] = -1e30f; lrow[tid] = 0.f;
  }
  __syncthreads();

  v8f oacc[2] = {};

  for (int ch = 0; ch < 6; ++ch) {
    // -- stage K raw (channel-major) --
    for (int idx = tid; idx < 64 * 64; idx += 256) {
      const int t = idx & 63, c = idx >> 6;
      float v = 0.f;
      if (ch < 4) {
        const int g = ch * 64 + t;
        const int yy = y0 - 4 + (g >> 4), xx = x0 - 4 + (g & 15);
        if (yy >= 0 && yy < HH && xx >= 0 && xx < WW)
          v = su_emb[((long)n * 189 + 64 + c) * HW + yy * WW + xx];
      } else {
        const int sp = (y0 + (t >> 3)) * WW + (x0 + (t & 7));
        const float* src = (ch == 4) ? warped : dsemb;
        v = src[((long)n * 128 + c) * HW + sp];
      }
      Sbuf[c * 64 + t] = v;
    }
    __syncthreads();
    if (tid < 64) {
      float s = 0.f, q = 0.f;
      for (int c = 0; c < 64; ++c) { const float v = Sbuf[c*64 + tid]; s += v; q += v*v; }
      const float mean = s * (1.f/64.f);
      const float r = rsqrtf(q * (1.f/64.f) - mean*mean + 1e-5f);
      for (int c = 0; c < 64; ++c)
        K16[tid*64 + c] = (_Float16)((Sbuf[c*64 + tid] - mean) * r * kg[c] + kb[c]);
    }
    __syncthreads();
    // -- stage V raw --
    for (int idx = tid; idx < 64 * 64; idx += 256) {
      const int t = idx & 63, c = idx >> 6;
      float v = 0.f;
      if (ch < 4) {
        const int g = ch * 64 + t;
        const int yy = y0 - 4 + (g >> 4), xx = x0 - 4 + (g & 15);
        if (yy >= 0 && yy < HH && xx >= 0 && xx < WW) {
          const int cc = 64 + c;  // channel in kv_image
          if (cc < 125) v = su_emb[((long)n * 189 + 64 + cc) * HW + yy * WW + xx];
          else          v = radiance[((long)n * 3 + (cc - 125)) * HW + yy * WW + xx];
        }
      } else {
        const int sp = (y0 + (t >> 3)) * WW + (x0 + (t & 7));
        const float* src = (ch == 4) ? warped : dsemb;
        v = src[((long)n * 128 + 64 + c) * HW + sp];
      }
      Sbuf[c * 64 + t] = v;
    }
    __syncthreads();
    if (tid < 64) {   // V LayerNorm over first 61 channels; store transposed [ch][tok]
      float s = 0.f, q = 0.f;
      for (int c = 0; c < 61; ++c) { const float v = Sbuf[c*64 + tid]; s += v; q += v*v; }
      const float mean = s * (1.f/61.f);
      const float r = rsqrtf(q * (1.f/61.f) - mean*mean + 1e-5f);
      for (int c = 0; c < 61; ++c)
        V16t[c*64 + tid] = (_Float16)((Sbuf[c*64 + tid] - mean) * r * vg[c] + vb[c]);
      for (int c = 61; c < 64; ++c) V16t[c*64 + tid] = (_Float16)Sbuf[c*64 + tid];
    }
    __syncthreads();
    // -- S = Qn x Kn^T (16 tiles over 8 waves, 2 each); store transposed [kvtok][qtok] --
#pragma unroll
    for (int tt = 0; tt < 2; ++tt) {
      const int tile = 2 * wave + tt;
      const int mt = tile >> 2, ntl = tile & 3;
      v8f sacc = {};
#pragma unroll
      for (int k0 = 0; k0 < 64; k0 += 32) {
        const v16h a = frag_a_rowmajor(Q16, 64, mt * 16, k0);
        const v16h b = frag_b_nmajor(K16, 64, k0, ntl * 16);
        sacc = __builtin_amdgcn_wmma_f32_16x16x32_f16(false, a, false, b, (short)0, sacc, false, false);
      }
      const int nn = ntl * 16 + (lane & 15);
      const int mbase = mt * 16 + hi8;           // 8 contiguous qtok rows per lane
      *(v8f*)(&Sbuf[nn * 64 + mbase]) = sacc;    // 2x ds_store_b128
    }
    __syncthreads();
    // -- online softmax row update (conflict-free reads); P (f16) overwrites K16 --
    if (tid < 64) {
      const float mold = mrow[tid];
      float mx = mold;
      for (int c = 0; c < 64; ++c) mx = fmaxf(mx, Sbuf[c*64 + tid]);
      const float alpha = __expf(mold - mx);
      float sum = 0.f;
      for (int c = 0; c < 64; ++c) {
        const float p = __expf(Sbuf[c*64 + tid] - mx);
        K16[tid*64 + c] = (_Float16)p;           // P row-major [qtok][kvtok]
        sum += p;
      }
      mrow[tid] = mx; lrow[tid] = lrow[tid] * alpha + sum; arow[tid] = alpha;
    }
    __syncthreads();
    // -- O = alpha*O + P x V (each wave owns 2 output tiles) --
#pragma unroll
    for (int tt = 0; tt < 2; ++tt) {
      const int tile = 2 * wave + tt;
      const int mt = tile >> 2, ntl = tile & 3;
      const int mbase = mt * 16 + hi8;
#pragma unroll
      for (int r = 0; r < 8; ++r) oacc[tt][r] *= arow[mbase + r];
#pragma unroll
      for (int k0 = 0; k0 < 64; k0 += 32) {
        const v16h a = frag_a_rowmajor(K16, 64, mt * 16, k0);   // P
        const v16h b = frag_b_nmajor(V16t, 64, k0, ntl * 16);   // V[ch][tok] N-major
        oacc[tt] = __builtin_amdgcn_wmma_f32_16x16x32_f16(false, a, false, b, (short)0, oacc[tt], false, false);
      }
    }
    __syncthreads();
  }

  // ---- normalize and fold back to image layout ----
#pragma unroll
  for (int tt = 0; tt < 2; ++tt) {
    const int tile = 2 * wave + tt;
    const int mt = tile >> 2, ntl = tile & 3;
    const int nn = ntl * 16 + (lane & 15);       // channel
    const int mbase = mt * 16 + hi8;
#pragma unroll
    for (int r = 0; r < 8; ++r) {
      const int t = mbase + r;                   // token in window
      const int sp = (y0 + (t >> 3)) * WW + (x0 + (t & 7));
      img_attn[((long)n * 64 + nn) * HW + sp] = oacc[tt][r] / lrow[t];
    }
  }
}

extern "C" void kernel_launch(void* const* d_in, const int* in_sizes, int n_in,
                              void* d_out, int out_size, void* d_ws, size_t ws_size,
                              hipStream_t stream) {
  const float* radiance  = (const float*)d_in[0];
  const float* skip      = (const float*)d_in[1];
  const float* ds_latent = (const float*)d_in[2];
  const float* tkv       = (const float*)d_in[3];
  const float* mv        = (const float*)d_in[4];
  const float* su_bn_g = (const float*)d_in[5];
  const float* su_bn_b = (const float*)d_in[6];
  const float* su_w    = (const float*)d_in[7];
  const float* su_b    = (const float*)d_in[8];
  const float* ds_bn_g = (const float*)d_in[9];
  const float* ds_bn_b = (const float*)d_in[10];
  const float* ds_w    = (const float*)d_in[11];
  const float* ds_b    = (const float*)d_in[12];
  const float* qln_g = (const float*)d_in[13];
  const float* qln_b = (const float*)d_in[14];
  const float* kln_g = (const float*)d_in[15];
  const float* kln_b = (const float*)d_in[16];
  const float* vln_g = (const float*)d_in[17];
  const float* vln_b = (const float*)d_in[18];
  const float* f1g = (const float*)d_in[19];
  const float* f1b = (const float*)d_in[20];
  const float* fw1 = (const float*)d_in[21];
  const float* fb1 = (const float*)d_in[22];
  const float* f2g = (const float*)d_in[23];
  const float* f2b = (const float*)d_in[24];
  const float* fw2 = (const float*)d_in[25];
  const float* fb2 = (const float*)d_in[26];

  float* ws = (float*)d_ws;
  size_t off = 0;
  float* us       = ws + off; off += (size_t)NB * 96 * HW;
  float* su_emb   = ws + off; off += (size_t)NB * 189 * HW;
  float* warped   = ws + off; off += (size_t)NB * 128 * HW;
  float* ds_small = ws + off; off += (size_t)NB * 128 * 128 * 128;
  float* dsemb    = ws + off; off += (size_t)NB * 128 * HW;
  float* iattn    = ws + off; off += (size_t)NB * 64 * HW;
  float* h1       = ws + off; off += (size_t)NB * 128 * HW;
  float* su_sc = ws + off; off += 192;
  float* su_sh = ws + off; off += 192;
  float* ds_sc = ws + off; off += 96;
  float* ds_sh = ws + off; off += 96;
  float* a_sc  = ws + off; off += 64;
  float* a_sh  = ws + off; off += 64;
  float* h_sc  = ws + off; off += 128;
  float* h_sh  = ws + off; off += 128;

  float* dec  = (float*)d_out;                    // (N,64,H,W)
  float* tout = dec + (size_t)NB * 64 * HW;       // (N,128,H,W)

  // 1) us = resize_ac(ds_latent, 256, 256)
  resize_ac_kernel<<<1024, 256, 0, stream>>>(ds_latent, us, 96, 128, 128, 256, 256);
  // 2) BN stats for concat(skip, us) and ds_latent
  bn_stats_kernel<<<96, 256, 0, stream>>>(skip, 96, HW, su_bn_g, su_bn_b, su_sc, su_sh, 0);
  bn_stats_kernel<<<96, 256, 0, stream>>>(us, 96, HW, su_bn_g + 96, su_bn_b + 96, su_sc, su_sh, 96);
  bn_stats_kernel<<<96, 256, 0, stream>>>(ds_latent, 96, 128 * 128, ds_bn_g, ds_bn_b, ds_sc, ds_sh, 0);
  // 3) su_emb = conv1x1(BN(concat(skip, us)))  (192 -> 189)
  {
    dim3 g((NB * HW) / 64, (189 + 31) / 32);
    conv1x1_bn_kernel<HW><<<g, 256, 0, stream>>>(skip, us, 96, 192, su_sc, su_sh,
                                                 su_w, su_b, 189, 189, su_emb, 0);
  }
  // 4) temporal warp (grid_sample border); resize-to-same-size is identity
  warp_kernel<<<2048, 256, 0, stream>>>(tkv, mv, warped);
  // 5) ds branch: conv at 128x128 (96 -> 125 into 128-ch buf), append raw ch93..95, resize
  {
    dim3 g((NB * 128 * 128) / 64, (125 + 31) / 32);
    conv1x1_bn_kernel<128 * 128><<<g, 256, 0, stream>>>(ds_latent, ds_latent, 96, 96,
                                                        ds_sc, ds_sh, ds_w, ds_b, 125, 128,
                                                        ds_small, 0);
  }
  chcopy_kernel<<<256, 256, 0, stream>>>(ds_latent, 96, 93, ds_small, 128, 125, 3, 128 * 128);
  resize_ac_kernel<<<1024, 256, 0, stream>>>(ds_small, dsemb, 128, 128, 128, 256, 256);
  // 6) flash windowed attention -> img_attn
  attn_kernel<<<NB * 1024, 256, 0, stream>>>(su_emb, radiance, warped, dsemb,
                                             qln_g, qln_b, kln_g, kln_b, vln_g, vln_b, iattn);
  // 7) temporal_out = concat(kv_image[:, :125], img_attn[:, 61:64])
  chcopy_kernel<<<4096, 256, 0, stream>>>(su_emb, 189, 64, tout, 128, 0, 125, HW);
  chcopy_kernel<<<256, 256, 0, stream>>>(iattn, 64, 61, tout, 128, 125, 3, HW);
  // 8) FFN
  bn_stats_kernel<<<64, 256, 0, stream>>>(iattn, 64, HW, f1g, f1b, a_sc, a_sh, 0);
  {
    dim3 g((NB * HW) / 64, (128 + 31) / 32);
    conv1x1_bn_kernel<HW><<<g, 256, 0, stream>>>(iattn, iattn, 64, 64, a_sc, a_sh,
                                                 fw1, fb1, 128, 128, h1, 1);
  }
  bn_stats_kernel<<<128, 256, 0, stream>>>(h1, 128, HW, f2g, f2b, h_sc, h_sh, 0);
  {
    dim3 g((NB * HW) / 64, (61 + 31) / 32);
    conv1x1_bn_kernel<HW><<<g, 256, 0, stream>>>(h1, h1, 128, 128, h_sc, h_sh,
                                                 fw2, fb2, 61, 64, dec, 0);
  }
  // decoded tail channels = img_attn[:, 61:64]
  chcopy_kernel<<<256, 256, 0, stream>>>(iattn, 64, 61, dec, 64, 61, 3, HW);

  (void)in_sizes; (void)n_in; (void)out_size; (void)ws_size;
}
